// LinformerAttention_48859547959765
// MI455X (gfx1250) — matrix operand order repaired and checked by reading, not compiled, not run on hardware
//
#include <hip/hip_runtime.h>
#include <hip/hip_bf16.h>

// Linformer attention for MI455X (gfx1250, wave32, WMMA f32_16x16x32_f16).
// Stage 1: Kp = (E*mask) @ K, VpT = ((E*mask) @ V)^T   (f16 WMMA, fp32 accum)
// Stage 2: softmax(Q @ Kp^T / 8) @ Vp                  (f16 WMMA, fp32 softmax)
// All WMMA fragments are assembled with 16-byte ds_load_b128; kernel 2's bulk
// tile staging uses the Tensor Data Mover (tensor_load_to_lds + TENSORcnt)
// when the builtin is available, with padded LDS row strides produced by the
// TDM pad_interval/pad_amount fields.

typedef __attribute__((ext_vector_type(4)))  float    f32x4;
typedef __attribute__((ext_vector_type(8)))  float    v8f;
typedef __attribute__((ext_vector_type(4)))  _Float16 v4h;
typedef __attribute__((ext_vector_type(8)))  _Float16 v8h;
typedef __attribute__((ext_vector_type(16))) _Float16 v16h;

constexpr int B_  = 2;
constexpr int H_  = 12;
constexpr int S_  = 4096;
constexpr int D_  = 64;
constexpr int LK_ = 256;

__device__ __forceinline__ v16h cat8(v8h lo, v8h hi) {
    return __builtin_shufflevector(lo, hi, 0, 1, 2, 3, 4, 5, 6, 7,
                                           8, 9, 10, 11, 12, 13, 14, 15);
}

#if __has_builtin(__builtin_amdgcn_tensor_load_to_lds) && \
    __has_builtin(__builtin_amdgcn_s_wait_tensorcnt)
#define USE_TDM 1
typedef __attribute__((ext_vector_type(4))) unsigned int u32x4;
typedef __attribute__((ext_vector_type(8))) int          i32x8;
typedef __attribute__((ext_vector_type(4))) int          i32x4;

// 2-D f16 tile DMA: global (rows x cols, contiguous) -> LDS at lds_byte_off,
// inserting pad_amt_dw DWORDs of padding after every (1<<pad_int_code) 8-byte
// chunks (i.e. after each row when sized to the row length).
__device__ __forceinline__ void tdm_load_tile_f16(const void* gaddr,
                                                  unsigned lds_byte_off,
                                                  unsigned cols, unsigned rows,
                                                  unsigned pad_int_code,
                                                  unsigned pad_amt_code) {
    const unsigned long long ga = (unsigned long long)gaddr;
    u32x4 g0;
    g0[0] = 1u;                                            // count=1, user D#
    g0[1] = lds_byte_off;                                  // lds_addr
    g0[2] = (unsigned)(ga & 0xffffffffu);                  // global_addr[31:0]
    g0[3] = (unsigned)((ga >> 32) & 0x1ffffffu)            // global_addr[56:32]
          | 0x80000000u;                                   // type = 2 (image)
    i32x8 g1;
    g1[0] = (int)((1u << 16)                               // data_size = 2B
                | (1u << 20)                               // pad_enable
                | (pad_int_code << 22)                     // pad_interval
                | (pad_amt_code << 25));                   // pad_amount
    g1[1] = (int)((cols & 0xffffu) << 16);                 // tensor_dim0[15:0]
    g1[2] = (int)((cols >> 16) | ((rows & 0xffffu) << 16));// dim0 hi | dim1 lo
    g1[3] = (int)((rows >> 16) | ((cols & 0xffffu) << 16));// dim1 hi | tile_dim0
    g1[4] = (int)(rows & 0xffffu);                         // tile_dim1 (tile_dim2=0)
    g1[5] = (int)cols;                                     // tensor_dim0_stride lo
    g1[6] = 0;                                             // stride hi | dim1_stride lo
    g1[7] = 0;
    const i32x4 z4 = {};
#if __clang_major__ >= 23
    const i32x8 z8 = {};
    __builtin_amdgcn_tensor_load_to_lds(g0, g1, z4, z4, z8, 0);
#else
    __builtin_amdgcn_tensor_load_to_lds(g0, g1, z4, z4, 0);
#endif
}
#else
#define USE_TDM 0
#endif

// ---------------------------------------------------------------------------
// Kernel 1: projection.  grid = (LK/32, H, B), block = 256 (8 waves).
// Block owns a 32(k) x 64(d) tile of Kp and Vp, looping S in 32-chunks.
// Wave w: ks = w>>2 -> 16 k-rows, dt = w&3 -> 16 d-cols.
// LDS: E*mask [32k][32s] stride 40;  K^T,V^T [64d][32s] stride 40.
// ---------------------------------------------------------------------------
__global__ __launch_bounds__(256)
void linformer_proj(const float* __restrict__ K, const float* __restrict__ V,
                    const float* __restrict__ mask, const float* __restrict__ E,
                    _Float16* __restrict__ Kp, _Float16* __restrict__ VpT) {
    __shared__ _Float16 ldsE[32 * 40];    // (E*mask)[k][s]
    __shared__ _Float16 ldsKT[64 * 40];   // K^T[d][s]
    __shared__ _Float16 ldsVT[64 * 40];   // V^T[d][s]

    const int tid  = threadIdx.x;
    const int lane = tid & 31;
    const int wave = tid >> 5;
    const int ks   = wave >> 2;           // 0..1
    const int dt   = wave & 3;            // 0..3
    const int mrow = lane & 15;
    const int half = lane >> 4;

    const int k0 = blockIdx.x * 32;
    const int h  = blockIdx.y;
    const int b  = blockIdx.z;

    const float* Eb = E + ((size_t)h * LK_ + k0) * S_;
    const float* Kb = K + ((size_t)b * H_ + h) * S_ * D_;
    const float* Vb = V + ((size_t)b * H_ + h) * S_ * D_;
    const float* mb = mask + (size_t)b * S_;

    v8f accK = {};
    v8f accV = {};

    // staging assignments (fixed per thread)
    const int erow = tid >> 3;            // 0..31 (k row)
    const int ecol = (tid & 7) * 4;       // 0..28 (s col)
    const int dcol = tid & 63;            // 0..63 (d)
    const int sg   = tid >> 6;            // 0..3  (8-s group)

    for (int s = 0; s < S_; s += 32) {
        // ---- stage E * mask (32k x 32s) : vector load, v4h store ----
        {
            const f32x4 e4 = *(const f32x4*)(Eb + (size_t)erow * S_ + s + ecol);
            const f32x4 m4 = *(const f32x4*)(mb + s + ecol);
            *(v4h*)&ldsE[erow * 40 + ecol] = __builtin_convertvector(e4 * m4, v4h);
        }
        // ---- stage K,V transposed: 8 along-s scalars -> one b128 store ----
        {
            const float* pk = Kb + (size_t)(s + sg * 8) * D_ + dcol;
            const float* pv = Vb + (size_t)(s + sg * 8) * D_ + dcol;
            if (s + 32 < S_) {
                __builtin_prefetch(pk + 32 * D_, 0, 1);
                __builtin_prefetch(pv + 32 * D_, 0, 1);
            }
            v8h hk, hv;
            #pragma unroll
            for (int j = 0; j < 8; ++j) {
                hk[j] = (_Float16)pk[(size_t)j * D_];
                hv[j] = (_Float16)pv[(size_t)j * D_];
            }
            *(v8h*)&ldsKT[dcol * 40 + sg * 8] = hk;
            *(v8h*)&ldsVT[dcol * 40 + sg * 8] = hv;
        }
        __syncthreads();

        // ---- fragments: all 16B ds_load_b128 ----
        const _Float16* ea = &ldsE[(ks * 16 + mrow) * 40 + half * 8];
        const v16h a = cat8(*(const v8h*)ea, *(const v8h*)(ea + 16));

        const _Float16* pkt = &ldsKT[(dt * 16 + mrow) * 40 + half * 8];
        const v16h bk = cat8(*(const v8h*)pkt, *(const v8h*)(pkt + 16));
        const _Float16* pvt = &ldsVT[(dt * 16 + mrow) * 40 + half * 8];
        const v16h bv = cat8(*(const v8h*)pvt, *(const v8h*)(pvt + 16));

        accK = __builtin_amdgcn_wmma_f32_16x16x32_f16(false, a, false, bk,
                                                      (short)0, accK, false, false);
        accV = __builtin_amdgcn_wmma_f32_16x16x32_f16(false, a, false, bv,
                                                      (short)0, accV, false, false);
        __syncthreads();
    }

    // ---- store: Kp row-major [k][d], Vp transposed [d][k], both f16 ----
    const size_t bh = (size_t)b * H_ + h;
    #pragma unroll
    for (int r = 0; r < 8; ++r) {
        const int M = r + half * 8;                     // C layout: M = r + 8*half
        const int kk = k0 + ks * 16 + M;
        const int dd = dt * 16 + mrow;                  // N = lane%16
        Kp[(bh * LK_ + kk) * D_ + dd]  = (_Float16)accK[r];
        VpT[(bh * D_ + dd) * LK_ + kk] = (_Float16)accV[r];
    }
}

// ---------------------------------------------------------------------------
// Kernel 2: attention.  grid = (S/64, H, B), block = 128 (4 waves).
// Each wave owns 16 query rows.  LDS: one tile buffer (Kp [256k][64d] stride 72,
// recycled as Vp^T [64d][256k] stride 264) + per-wave 16x256 f16 prob rows.
// Tile staging via TDM (tensor_load_to_lds) when available.
// ---------------------------------------------------------------------------
__global__ __launch_bounds__(128)
void linformer_attn(const float* __restrict__ Q,
                    const _Float16* __restrict__ Kp,
                    const _Float16* __restrict__ VpT,
                    float* __restrict__ out) {
    extern __shared__ char smem[];
    _Float16* tiles = (_Float16*)smem;            // 256*72 halves (Kp, then Vp^T)
    _Float16* probs = tiles + 256 * 72;           // 4 waves * 16 * 256 halves

    const int tid  = threadIdx.x;
    const int lane = tid & 31;
    const int wave = tid >> 5;
    const int mrow = lane & 15;
    const int half = lane >> 4;

    const int h = blockIdx.y;
    const int b = blockIdx.z;
    const size_t bh = (size_t)b * H_ + h;
    const int q0 = blockIdx.x * 64 + wave * 16;

    // ---- stage Kp [256][64] -> LDS rows padded to stride 72 ----
#if USE_TDM
    if (wave == 0) {
        // pad after every 32 DWORDs (one 64-half row): code 4; pad 4 DWORDs: code 3
        tdm_load_tile_f16(Kp + bh * LK_ * D_, 0u, (unsigned)D_, (unsigned)LK_, 4u, 3u);
        __builtin_amdgcn_s_wait_tensorcnt(0);
    }
#else
    {
        const uint4* src = (const uint4*)(Kp + bh * LK_ * D_);
        for (int idx = tid; idx < LK_ * D_ / 8; idx += 128) {   // 2048 uint4
            const int row = idx >> 3;                            // 8 uint4 / row
            const int c8  = idx & 7;
            *(uint4*)&tiles[row * 72 + c8 * 8] = src[idx];
        }
    }
#endif
    __syncthreads();

    // ---- Q fragments (vector f32 loads + pack-convert), reused 16x ----
    v16h qf0, qf1;
    {
        const float* Qrow = Q + (bh * S_ + q0 + mrow) * D_;
        const v8f qa = *(const v8f*)(Qrow + half * 8);
        const v8f qb = *(const v8f*)(Qrow + 16 + half * 8);
        const v8f qc = *(const v8f*)(Qrow + 32 + half * 8);
        const v8f qd = *(const v8f*)(Qrow + 48 + half * 8);
        qf0 = cat8(__builtin_convertvector(qa, v8h), __builtin_convertvector(qb, v8h));
        qf1 = cat8(__builtin_convertvector(qc, v8h), __builtin_convertvector(qd, v8h));
    }

    _Float16* myp = probs + wave * 16 * 256;

    // ---- scores: 16 k-tiles, D split into two K=32 WMMA steps ----
    for (int t = 0; t < 16; ++t) {
        const _Float16* kr = &tiles[(t * 16 + mrow) * 72 + half * 8];
        const v16h b0 = cat8(*(const v8h*)kr,        *(const v8h*)(kr + 16));
        const v16h b1 = cat8(*(const v8h*)(kr + 32), *(const v8h*)(kr + 48));
        v8f acc = {};
        acc = __builtin_amdgcn_wmma_f32_16x16x32_f16(false, qf0, false, b0,
                                                     (short)0, acc, false, false);
        acc = __builtin_amdgcn_wmma_f32_16x16x32_f16(false, qf1, false, b1,
                                                     (short)0, acc, false, false);
        #pragma unroll
        for (int r = 0; r < 8; ++r) {
            const int M = r + half * 8;
            myp[M * 256 + t * 16 + mrow] = (_Float16)(acc[r] * 0.125f);
        }
    }

    // ---- fp32 softmax (lane owns 8 contiguous k: one b128 load/store) ----
    for (int m = 0; m < 16; ++m) {
        _Float16* prow = &myp[m * 256 + lane * 8];
        v8f v = __builtin_convertvector(*(const v8h*)prow, v8f);
        float mx = -3.0e38f;
        #pragma unroll
        for (int j = 0; j < 8; ++j) mx = fmaxf(mx, v[j]);
        #pragma unroll
        for (int o = 16; o > 0; o >>= 1) mx = fmaxf(mx, __shfl_xor(mx, o, 32));
        float sum = 0.0f;
        #pragma unroll
        for (int j = 0; j < 8; ++j) { v[j] = __expf(v[j] - mx); sum += v[j]; }
        #pragma unroll
        for (int o = 16; o > 0; o >>= 1) sum += __shfl_xor(sum, o, 32);
        const float inv = 1.0f / sum;
        *(v8h*)prow = __builtin_convertvector(v * inv, v8h);
    }

    // ---- recycle tile buffer: Kp -> Vp^T [64][256] stride 264 ----
    __syncthreads();
#if USE_TDM
    if (wave == 0) {
        // pad after every 128 DWORDs (one 256-half row): code 6; pad 4 DWORDs: code 3
        tdm_load_tile_f16(VpT + bh * D_ * LK_, 0u, (unsigned)LK_, (unsigned)D_, 6u, 3u);
        __builtin_amdgcn_s_wait_tensorcnt(0);
    }
#else
    {
        const uint4* src = (const uint4*)(VpT + bh * D_ * LK_);
        for (int idx = tid; idx < D_ * LK_ / 8; idx += 128) {    // 2048 uint4
            const int row = idx >> 5;                             // 32 uint4 / row
            const int c8  = idx & 31;
            *(uint4*)&tiles[row * 264 + c8 * 8] = src[idx];
        }
    }
#endif
    __syncthreads();

    // ---- out = P (16x256) @ Vp (256x64): 8 K=32 chunks, 4 d-tiles ----
    v8f accO[4] = {{}, {}, {}, {}};
    for (int c = 0; c < 8; ++c) {
        const _Float16* pa = &myp[mrow * 256 + c * 32 + half * 8];
        const v16h a = cat8(*(const v8h*)pa, *(const v8h*)(pa + 16));
        #pragma unroll
        for (int dt = 0; dt < 4; ++dt) {
            const _Float16* pb = &tiles[(dt * 16 + mrow) * 264 + c * 32 + half * 8];
            const v16h bb = cat8(*(const v8h*)pb, *(const v8h*)(pb + 16));
            accO[dt] = __builtin_amdgcn_wmma_f32_16x16x32_f16(false, a, false, bb,
                                                              (short)0, accO[dt],
                                                              false, false);
        }
    }

    // ---- write fp32 output ----
    #pragma unroll
    for (int dt = 0; dt < 4; ++dt)
        #pragma unroll
        for (int r = 0; r < 8; ++r) {
            const int M = r + half * 8;
            out[(bh * S_ + q0 + M) * D_ + dt * 16 + mrow] = accO[dt][r];
        }
}

// ---------------------------------------------------------------------------
extern "C" void kernel_launch(void* const* d_in, const int* in_sizes, int n_in,
                              void* d_out, int out_size, void* d_ws, size_t ws_size,
                              hipStream_t stream) {
    (void)in_sizes; (void)n_in; (void)out_size; (void)ws_size;
    const float* Q    = (const float*)d_in[0];
    const float* K    = (const float*)d_in[1];
    const float* V    = (const float*)d_in[2];
    const float* mask = (const float*)d_in[3];
    const float* E    = (const float*)d_in[4];
    float* out = (float*)d_out;

    _Float16* Kp  = (_Float16*)d_ws;                      // [B,H,LK,D] f16
    _Float16* VpT = Kp + (size_t)B_ * H_ * LK_ * D_;      // [B,H,D,LK] f16

    linformer_proj<<<dim3(LK_ / 32, H_, B_), 256, 0, stream>>>(K, V, mask, E, Kp, VpT);

    const size_t smem = (256 * 72 + 4 * 16 * 256) * sizeof(_Float16); // 69632 B
    linformer_attn<<<dim3(S_ / 64, H_, B_), 128, smem, stream>>>(Q, Kp, VpT, out);
}
